// Decoder_59193239273883
// MI455X (gfx1250) — matrix operand order repaired and checked by reading, not compiled, not run on hardware
//
#include <hip/hip_runtime.h>
#include <hip/hip_bf16.h>

// ---------------- types ----------------
typedef __attribute__((ext_vector_type(16))) __bf16 v16bf;
typedef __attribute__((ext_vector_type(8)))  __bf16 v8bf;
typedef __attribute__((ext_vector_type(8)))  float  v8f;

#define Bq   32
#define Sq   256
#define Tq   128
#define Hq   512
#define EMBq 256
#define ENCq 1024
#define EDq  512
#define G4H  2048            // 4*H
#define KG   2304            // 1792 + 512 : [emb|ctx|edit|h]
#define KPRE 1792            // [emb|h|ctx]
#define BTH  (Bq*Tq*Hq)      // 2097152
#define BH   (Bq*Hq)         // 16384

// ---------------- helpers ----------------
__device__ __forceinline__ __bf16 f2bf(float f) {
    union { float f; unsigned u; } v; v.f = f;
    unsigned r = v.u + 0x7FFFu + ((v.u >> 16) & 1u);
    unsigned short s = (unsigned short)(r >> 16);
    __bf16 o; __builtin_memcpy(&o, &s, 2); return o;
}
__device__ __forceinline__ float bf2f(__bf16 b) {
    unsigned short s; __builtin_memcpy(&s, &b, 2);
    union { unsigned u; float f; } v; v.u = ((unsigned)s) << 16;
    return v.f;
}
__device__ __forceinline__ float sigm(float x) { return 1.f / (1.f + __expf(-x)); }

// ---------------- WMMA bf16 GEMM:  C[M,N] = A[M,K] * B[N,K]^T (+bias[n]) ----------------
// One wave computes a 16 x (16*NT) output strip: A fragment is loaded once per K-slab and
// reused across NT WMMAs (NT independent accumulators hide WMMA latency).
// A,B row-major with K contiguous (lda/ldb elems). ntn counts strips of NT tiles.
template<int NT>
__global__ void k_wmma_gemm(const __bf16* __restrict__ A, int lda,
                            const __bf16* __restrict__ Bm, int ldb,
                            const float* __restrict__ bias,
                            float* __restrict__ C, int ldc,
                            int ntm, int ntn, int K)
{
    int wave = (blockIdx.x * blockDim.x + threadIdx.x) >> 5;
    int lane = threadIdx.x & 31;
    int tm = wave / ntn, tn = wave - tm * ntn;
    if (tm >= ntm) return;                 // wave-uniform: EXEC stays all-1 for WMMA
    int m0 = tm << 4, n0 = tn * (16 * NT);

    const __bf16* pA = A + (size_t)(m0 + (lane & 15)) * lda + ((lane >> 4) * 16);
    const __bf16* pB[NT];
#pragma unroll
    for (int nt = 0; nt < NT; ++nt)
        pB[nt] = Bm + (size_t)(n0 + nt * 16 + (lane & 15)) * ldb + ((lane >> 4) * 16);

    v8f acc[NT];
#pragma unroll
    for (int nt = 0; nt < NT; ++nt) acc[nt] = (v8f){};

    for (int k = 0; k < K; k += 32) {
        __builtin_prefetch(pA + k + 64, 0, 1);
        v8bf a_lo = *reinterpret_cast<const v8bf*>(pA + k);
        v8bf a_hi = *reinterpret_cast<const v8bf*>(pA + k + 8);
        v16bf afrag;
#pragma unroll
        for (int i = 0; i < 8; ++i) { afrag[i] = a_lo[i]; afrag[i + 8] = a_hi[i]; }
#pragma unroll
        for (int nt = 0; nt < NT; ++nt) {
            __builtin_prefetch(pB[nt] + k + 64, 0, 1);
            v16bf bfrag = *reinterpret_cast<const v16bf*>(pB[nt] + k);
            acc[nt] = __builtin_amdgcn_wmma_f32_16x16x32_bf16(
                          false, afrag, false, bfrag, (short)0, acc[nt], false, false);
        }
    }
    int mh = (lane >> 4) * 8;
#pragma unroll
    for (int nt = 0; nt < NT; ++nt) {
        int n = n0 + nt * 16 + (lane & 15);
        float bv = bias ? bias[n] : 0.f;
#pragma unroll
        for (int j = 0; j < 8; ++j) {
            int m = m0 + j + mh;
            C[(size_t)m * ldc + n] = acc[nt][j] + bv;
        }
    }
}

// ---------------- setup conversion kernels ----------------
__global__ void k_f32_to_bf16(__bf16* __restrict__ dst, const float* __restrict__ src, int n) {
    for (int i = blockIdx.x * blockDim.x + threadIdx.x; i < n; i += gridDim.x * blockDim.x)
        dst[i] = f2bf(src[i]);
}
// Wg[n, 0:1792] = W_ih[n,:], Wg[n, 1792:2304] = W_hh[n,:]
__global__ void k_build_Wg(__bf16* __restrict__ Wg, const float* __restrict__ Wih,
                           const float* __restrict__ Whh, int n) {
    for (int i = blockIdx.x * blockDim.x + threadIdx.x; i < n; i += gridDim.x * blockDim.x) {
        int r = i / KG, k = i - r * KG;
        float v = (k < 1792) ? Wih[r * 1792 + k] : Whh[r * Hq + (k - 1792)];
        Wg[i] = f2bf(v);
    }
}
// Xpre[(b*T+t), 0:256] = emb ; rows stride KPRE
__global__ void k_fill_xpre_emb(__bf16* __restrict__ xpre, const float* __restrict__ emb, int n) {
    for (int i = blockIdx.x * blockDim.x + threadIdx.x; i < n; i += gridDim.x * blockDim.x) {
        int bt = i / EMBq, e = i - bt * EMBq;
        xpre[(size_t)bt * KPRE + e] = f2bf(emb[i]);
    }
}
// xg[b, 1280:1792] = edit_last[b,:]
__global__ void k_fill_xg_edit(__bf16* __restrict__ xg, const float* __restrict__ edit, int n) {
    for (int i = blockIdx.x * blockDim.x + threadIdx.x; i < n; i += gridDim.x * blockDim.x) {
        int b = i / EDq, j = i - b * EDq;
        xg[b * KG + 1280 + j] = f2bf(edit[i]);
    }
}
__global__ void k_bias_sum(float* __restrict__ bsum, const float* __restrict__ bih,
                           const float* __restrict__ bhh, int n) {
    for (int i = blockIdx.x * blockDim.x + threadIdx.x; i < n; i += gridDim.x * blockDim.x)
        bsum[i] = bih[i] + bhh[i];
}

// ---------------- bridge: h = tanh([enc_h|edit_h] @ Wb.T + bb), same for c ---------------
__global__ void k_bridge(float* __restrict__ h, float* __restrict__ c, __bf16* __restrict__ xg,
                         const float* __restrict__ ench, const float* __restrict__ edith,
                         const float* __restrict__ encc, const float* __restrict__ editc,
                         const float* __restrict__ Wb, const float* __restrict__ bb) {
    int idx = blockIdx.x * blockDim.x + threadIdx.x;
    if (idx >= BH) return;
    int b = idx / Hq, hh = idx - b * Hq;
    const float* Wr = Wb + (size_t)hh * (ENCq + EDq);
    float aH = bb[hh], aC = bb[hh];
    const float* eh = ench + b * ENCq; const float* ec = encc + b * ENCq;
    for (int k = 0; k < ENCq; ++k) { float w = Wr[k]; aH += eh[k] * w; aC += ec[k] * w; }
    const float* dh = edith + b * EDq; const float* dc = editc + b * EDq;
    for (int k = 0; k < EDq; ++k) { float w = Wr[ENCq + k]; aH += dh[k] * w; aC += dc[k] * w; }
    float hv = tanhf(aH), cv = tanhf(aC);
    h[idx] = hv; c[idx] = cv;
    xg[b * KG + 1792 + hh] = f2bf(hv);
}

// ---------------- per-step: write emb_t into xg[:,0:256] ----------------
__global__ void k_emb_step(__bf16* __restrict__ xg, const float* __restrict__ emb, int t) {
    int i = blockIdx.x * blockDim.x + threadIdx.x;
    if (i >= Bq * EMBq) return;
    int b = i / EMBq, e = i - b * EMBq;
    xg[b * KG + e] = f2bf(emb[((size_t)b * Tq + t) * EMBq + e]);
}

// ---------------- attention: scores -> softmax -> context (one block per batch b) --------
__global__ void k_attention(const float* __restrict__ query, const float* __restrict__ pkey,
                            const float* __restrict__ ev, const unsigned char* __restrict__ mask,
                            const __bf16* __restrict__ encbf,
                            __bf16* __restrict__ xg, __bf16* __restrict__ xpre, int t)
{
    int b = blockIdx.x, s = threadIdx.x;   // blockDim = 256 = S
    __shared__ float prob[Sq];
    __shared__ float red[Sq];
    const float* q = query + b * Hq;
    const float* pk = pkey + ((size_t)b * Sq + s) * Hq;
    float sc = 0.f;
    for (int h = 0; h < Hq; ++h) sc += tanhf(q[h] + pk[h]) * ev[h];
    if (!mask[b * Sq + s]) sc = -1e9f;
    red[s] = sc; __syncthreads();
    for (int o = 128; o > 0; o >>= 1) { if (s < o) red[s] = fmaxf(red[s], red[s + o]); __syncthreads(); }
    float mx = red[0]; __syncthreads();
    float e = __expf(sc - mx);
    red[s] = e; __syncthreads();
    for (int o = 128; o > 0; o >>= 1) { if (s < o) red[s] += red[s + o]; __syncthreads(); }
    float inv = 1.f / red[0];
    prob[s] = e * inv; __syncthreads();
    // context: each thread owns 4 columns of ENC2
    const __bf16* eb = encbf + (size_t)b * Sq * ENCq;
#pragma unroll
    for (int r = 0; r < 4; ++r) {
        int d = s + (r << 8);
        float acc = 0.f;
        for (int ss = 0; ss < Sq; ++ss) acc += prob[ss] * bf2f(eb[(size_t)ss * ENCq + d]);
        __bf16 cb = f2bf(acc);
        xg[b * KG + EMBq + d] = cb;                               // gates input [ctx]
        xpre[((size_t)b * Tq + t) * KPRE + EMBq + Hq + d] = cb;   // pre input   [ctx]
    }
}

// ---------------- LSTM pointwise + state/output scatter ----------------
__global__ void k_lstm(const float* __restrict__ gates, float* __restrict__ c, float* __restrict__ h,
                       __bf16* __restrict__ xg, __bf16* __restrict__ xpre,
                       float* __restrict__ out, int t)
{
    int idx = blockIdx.x * blockDim.x + threadIdx.x;
    if (idx >= BH) return;
    int b = idx / Hq, hh = idx - b * Hq;
    const float* g = gates + (size_t)b * G4H;
    float gi = g[hh], gf = g[Hq + hh], gg = g[2 * Hq + hh], go = g[3 * Hq + hh];
    float cn = sigm(gf) * c[idx] + sigm(gi) * tanhf(gg);
    float hn = sigm(go) * tanhf(cn);
    c[idx] = cn; h[idx] = hn;
    out[((size_t)b * Tq + t) * Hq + hh] = hn;                     // decoder_states
    __bf16 hb = f2bf(hn);
    xg[b * KG + 1792 + hh] = hb;                                  // gates input [h]
    xpre[((size_t)b * Tq + t) * KPRE + EMBq + hh] = hb;           // pre input   [h]
}

__global__ void k_copy_hc(float* __restrict__ out, const float* __restrict__ h,
                          const float* __restrict__ c) {
    int i = blockIdx.x * blockDim.x + threadIdx.x;
    if (i >= BH) return;
    out[BTH + i] = h[i];
    out[BTH + BH + i] = c[i];
}

// ---------------- host ----------------
static inline size_t alnup(size_t x) { return (x + 255) & ~(size_t)255; }

extern "C" void kernel_launch(void* const* d_in, const int* in_sizes, int n_in,
                              void* d_out, int out_size, void* d_ws, size_t ws_size,
                              hipStream_t stream)
{
    const float* trg_embed  = (const float*)d_in[0];
    const float* edit_hid   = (const float*)d_in[1];
    const float* edit_cell  = (const float*)d_in[2];
    const float* enc_out    = (const float*)d_in[3];
    const float* enc_hid    = (const float*)d_in[4];
    const float* enc_cell   = (const float*)d_in[5];
    const unsigned char* src_mask = (const unsigned char*)d_in[6];
    const float* bridge_W   = (const float*)d_in[7];
    const float* bridge_b   = (const float*)d_in[8];
    const float* key_W      = (const float*)d_in[9];
    const float* query_W    = (const float*)d_in[10];
    const float* energy_v   = (const float*)d_in[11];
    const float* W_ih       = (const float*)d_in[12];
    const float* W_hh       = (const float*)d_in[13];
    const float* b_ih       = (const float*)d_in[14];
    const float* b_hh       = (const float*)d_in[15];
    const float* pre_W      = (const float*)d_in[16];
    float* out = (float*)d_out;

    // workspace layout (bytes)
    char* ws = (char*)d_ws;
    size_t off = 0;
    __bf16* Wg_bf   = (__bf16*)(ws + off); off = alnup(off + (size_t)G4H * KG * 2);
    __bf16* qW_bf   = (__bf16*)(ws + off); off = alnup(off + (size_t)Hq * Hq * 2);
    __bf16* kW_bf   = (__bf16*)(ws + off); off = alnup(off + (size_t)Hq * ENCq * 2);
    __bf16* pW_bf   = (__bf16*)(ws + off); off = alnup(off + (size_t)Hq * KPRE * 2);
    __bf16* enc_bf  = (__bf16*)(ws + off); off = alnup(off + (size_t)Bq * Sq * ENCq * 2);
    float*  pkey    = (float*)(ws + off);  off = alnup(off + (size_t)Bq * Sq * Hq * 4);
    __bf16* xpre    = (__bf16*)(ws + off); off = alnup(off + (size_t)Bq * Tq * KPRE * 2);
    __bf16* xg      = (__bf16*)(ws + off); off = alnup(off + (size_t)Bq * KG * 2);
    float*  query   = (float*)(ws + off);  off = alnup(off + (size_t)Bq * Hq * 4);
    float*  gates   = (float*)(ws + off);  off = alnup(off + (size_t)Bq * G4H * 4);
    float*  hstate  = (float*)(ws + off);  off = alnup(off + (size_t)BH * 4);
    float*  cstate  = (float*)(ws + off);  off = alnup(off + (size_t)BH * 4);
    float*  bsum    = (float*)(ws + off);  off = alnup(off + (size_t)G4H * 4);
    (void)ws_size; (void)n_in; (void)in_sizes; (void)out_size;

    auto gs = [](int n) { int b = (n + 255) / 256; return b > 4096 ? 4096 : b; };

    // ---- one-time setup (per call) ----
    k_build_Wg<<<gs(G4H * KG), 256, 0, stream>>>(Wg_bf, W_ih, W_hh, G4H * KG);
    k_f32_to_bf16<<<gs(Hq * Hq), 256, 0, stream>>>(qW_bf, query_W, Hq * Hq);
    k_f32_to_bf16<<<gs(Hq * ENCq), 256, 0, stream>>>(kW_bf, key_W, Hq * ENCq);
    k_f32_to_bf16<<<gs(Hq * KPRE), 256, 0, stream>>>(pW_bf, pre_W, Hq * KPRE);
    k_f32_to_bf16<<<gs(Bq * Sq * ENCq), 256, 0, stream>>>(enc_bf, enc_out, Bq * Sq * ENCq);
    k_fill_xpre_emb<<<gs(Bq * Tq * EMBq), 256, 0, stream>>>(xpre, trg_embed, Bq * Tq * EMBq);
    k_fill_xg_edit<<<gs(Bq * EDq), 256, 0, stream>>>(xg, edit_hid, Bq * EDq);
    k_bias_sum<<<gs(G4H), 256, 0, stream>>>(bsum, b_ih, b_hh, G4H);
    k_bridge<<<(BH + 255) / 256, 256, 0, stream>>>(hstate, cstate, xg, enc_hid, edit_hid,
                                                   enc_cell, edit_cell, bridge_W, bridge_b);
    // proj_key = enc_out @ key_W.T : (8192 x 512) K=1024 ; 16x64 strips per wave
    {
        int ntm = (Bq * Sq) / 16, ntn = Hq / 64;
        int blocks = (ntm * ntn + 7) / 8;
        k_wmma_gemm<4><<<blocks, 256, 0, stream>>>(enc_bf, ENCq, kW_bf, ENCq, nullptr,
                                                   pkey, Hq, ntm, ntn, ENCq);
    }

    // ---- sequential decode loop (latency-bound: max-parallel 16x16 tiles) ----
    for (int t = 0; t < Tq; ++t) {
        k_emb_step<<<(Bq * EMBq + 255) / 256, 256, 0, stream>>>(xg, trg_embed, t);
        // query = h @ query_W.T : (32 x 512) K=512 ; A = xg h-slice
        {
            int ntm = Bq / 16, ntn = Hq / 16;
            k_wmma_gemm<1><<<(ntm * ntn + 7) / 8, 256, 0, stream>>>(xg + 1792, KG, qW_bf, Hq,
                                                                    nullptr, query, Hq, ntm, ntn, Hq);
        }
        k_attention<<<Bq, Sq, 0, stream>>>(query, pkey, energy_v, src_mask, enc_bf, xg, xpre, t);
        // gates = [emb|ctx|edit|h] @ [W_ih|W_hh].T + (b_ih+b_hh) : (32 x 2048) K=2304
        {
            int ntm = Bq / 16, ntn = G4H / 16;
            k_wmma_gemm<1><<<(ntm * ntn + 7) / 8, 256, 0, stream>>>(xg, KG, Wg_bf, KG,
                                                                    bsum, gates, G4H, ntm, ntn, KG);
        }
        k_lstm<<<(BH + 255) / 256, 256, 0, stream>>>(gates, cstate, hstate, xg, xpre, out, t);
    }

    // ---- tail: final state + batched pre-output GEMM ----
    k_copy_hc<<<(BH + 255) / 256, 256, 0, stream>>>(out, hstate, cstate);
    // pre = [emb|h|ctx] @ pre_W.T : (4096 x 512) K=1792 ; 16x64 strips per wave
    {
        int ntm = (Bq * Tq) / 16, ntn = Hq / 64;
        int blocks = (ntm * ntn + 7) / 8;
        k_wmma_gemm<4><<<blocks, 256, 0, stream>>>(xpre, KPRE, pW_bf, KPRE, nullptr,
                                                   out + BTH + 2 * BH, Hq, ntm, ntn, KPRE);
    }
}
